// DeformableTransformer_31714038514179
// MI455X (gfx1250) — matrix operand order repaired and checked by reading, not compiled, not run on hardware
//
#include <hip/hip_runtime.h>
#include <math.h>

// ---------------------------------------------------------------------------
// Model constants (fixed by the reference)
// ---------------------------------------------------------------------------
#define NB   2
#define LENT 7681          // 76*76 + 38*38 + 19*19 + 10*10
#define NQQ  300
#define DD   256
#define NHH  8
#define HDD  32
#define DFFF 1024

typedef __bf16 bf16_t;
typedef bf16_t v16bf __attribute__((ext_vector_type(16)));
typedef float  v8f   __attribute__((ext_vector_type(8)));

#if defined(__has_builtin)
#if __has_builtin(__builtin_amdgcn_cvt_pk_bf16_f32)
#define HAVE_CVT_PK_BF16 1
#endif
#endif

__device__ __forceinline__ unsigned short f2bf(float f) {
    unsigned int u = __float_as_uint(f);
    u += 0x7FFFu + ((u >> 16) & 1u);          // round-to-nearest-even
    return (unsigned short)(u >> 16);
}
__device__ __forceinline__ unsigned int pack2bf(float lo, float hi) {
#ifdef HAVE_CVT_PK_BF16
    auto r = __builtin_amdgcn_cvt_pk_bf16_f32(lo, hi);   // v_cvt_pk_bf16_f32
    union { decltype(r) v; unsigned int u; } c;
    c.v = r;
    return c.u;
#else
    return (unsigned int)f2bf(lo) | ((unsigned int)f2bf(hi) << 16);
#endif
}

// ---------------------------------------------------------------------------
// WMMA GEMM:  C[M,N] = act(A[M,K] @ W[K,N] + bias)
// 256 threads = 8 waves; tile 128(M) x 64(N), BK = 32, double-buffered LDS.
// Branchless staging: clamped vector loads + select-zero, packed bf16 stores.
// All fragments prefetched to registers -> back-to-back WMMA issue.
// Requires N % 64 == 0, K % 32 == 0 (true for all layers here). M guarded.
// ---------------------------------------------------------------------------
__global__ __launch_bounds__(256)
void gemm_bf16(const float* __restrict__ A, int lda,
               const float* __restrict__ W, int ldw,
               const float* __restrict__ bias,
               float* __restrict__ C, int ldc,
               int M, int N, int K, int relu)
{
    const int LDS_S = 40;                        // 32 + 8 pad (halves)
    __shared__ unsigned short As[2][128 * 40];
    __shared__ unsigned short Bs[2][64 * 40];

    const int tid  = threadIdx.x;
    const int lane = tid & 31;
    const int wave = tid >> 5;
    const int m0 = blockIdx.y * 128;
    const int n0 = blockIdx.x * 64;

    v8f acc[4];
    #pragma unroll
    for (int nt = 0; nt < 4; ++nt)
        #pragma unroll
        for (int r = 0; r < 8; ++r) acc[nt][r] = 0.0f;

    const int kbA = (lane >= 16) ? 8 : 0;        // A-frag K sub-block base
    const int kbB = (lane >= 16) ? 16 : 0;       // unused; B frag also uses kbA
    const int rw  = lane & 15;

    // staging geometry
    const int amr = tid >> 3;                    // A row base (stride 32)
    const int akc = (tid & 7) * 4;               // A k offset (4 floats)
    const int bnc = tid & 63;                    // B n offset
    const int bkp = tid >> 6;                    // B k-pair base (stride 4)

    float4 ar[4];                                // staged A (4 x float4)
    float  br[8];                                // staged W (4 k-pairs)

    auto loadA = [&](int k0) {
        #pragma unroll
        for (int i = 0; i < 4; ++i) {
            int gm  = m0 + amr + 32 * i;
            int gmc = (gm < M) ? gm : (M - 1);
            ar[i] = *(const float4*)(A + (size_t)gmc * lda + (k0 + akc));
        }
    };
    auto loadB = [&](int k0) {
        #pragma unroll
        for (int i = 0; i < 4; ++i) {
            int kp = bkp + 4 * i;                // 0..15
            const float* wp = W + (size_t)(k0 + 2 * kp) * ldw + (n0 + bnc);
            br[2 * i]     = wp[0];
            br[2 * i + 1] = wp[ldw];
        }
    };
    auto storeA = [&](int buf) {
        #pragma unroll
        for (int i = 0; i < 4; ++i) {
            int m = amr + 32 * i;
            bool ok = (m0 + m) < M;
            float x = ok ? ar[i].x : 0.0f;
            float y = ok ? ar[i].y : 0.0f;
            float z = ok ? ar[i].z : 0.0f;
            float w = ok ? ar[i].w : 0.0f;
            uint2 p; p.x = pack2bf(x, y); p.y = pack2bf(z, w);
            *(uint2*)(&As[buf][m * LDS_S + akc]) = p;
        }
    };
    auto storeB = [&](int buf) {
        #pragma unroll
        for (int i = 0; i < 4; ++i) {
            int kp = bkp + 4 * i;
            *(unsigned int*)(&Bs[buf][bnc * LDS_S + 2 * kp]) =
                pack2bf(br[2 * i], br[2 * i + 1]);
        }
    };
    auto compute = [&](int buf) {
        union Frag { unsigned int u[8]; v16bf v; };
        // prefetch A fragment + all 4 B fragments, then issue WMMAs back-to-back
        Frag af;
        {
            const unsigned short* ap = &As[buf][(wave * 16 + rw) * LDS_S];
            const unsigned int* a0 = (const unsigned int*)(ap + kbA);
            const unsigned int* a1 = (const unsigned int*)(ap + 16 + kbA);
            #pragma unroll
            for (int j = 0; j < 4; ++j) { af.u[j] = a0[j]; af.u[4 + j] = a1[j]; }
        }
        Frag bg[4];
        #pragma unroll
        for (int nt = 0; nt < 4; ++nt) {
            const unsigned short* bp = &Bs[buf][(nt * 16 + rw) * LDS_S];
            const unsigned int* b0 = (const unsigned int*)(bp + kbA);
            const unsigned int* b1 = (const unsigned int*)(bp + 16 + kbA);
            #pragma unroll
            for (int j = 0; j < 4; ++j) { bg[nt].u[j] = b0[j]; bg[nt].u[4 + j] = b1[j]; }
        }
        #pragma unroll
        for (int nt = 0; nt < 4; ++nt)
            acc[nt] = __builtin_amdgcn_wmma_f32_16x16x32_bf16(
                false, af.v, false, bg[nt].v, (short)0, acc[nt], false, false);
    };

    const int nk = K >> 5;
    loadA(0); loadB(0);
    for (int kb = 0; kb < nk; ++kb) {
        int cur = kb & 1;
        storeA(cur); storeB(cur);                // consumes regs staged for tile kb
        __syncthreads();                         // store->compute; also WAR for buf[cur]
        if (kb + 1 < nk) {                       // issue next tile's global loads,
            loadA((kb + 1) << 5);                // overlapped with WMMA below
            loadB((kb + 1) << 5);
        }
        compute(cur);
    }

    // ---- epilogue: bias (+ReLU), C/D layout M = r + 8*(lane>=16), N = lane&15
    const int mhi = (lane >> 4) * 8;
    #pragma unroll
    for (int nt = 0; nt < 4; ++nt) {
        int n = n0 + nt * 16 + rw;
        float bv = bias ? bias[n] : 0.0f;
        #pragma unroll
        for (int r = 0; r < 8; ++r) {
            int m = m0 + wave * 16 + mhi + r;
            if (m < M) {
                float v = acc[nt][r] + bv;
                if (relu) v = fmaxf(v, 0.0f);
                C[(size_t)m * ldc + n] = v;
            }
        }
    }
}

// ---------------------------------------------------------------------------
// Elementwise helpers
// ---------------------------------------------------------------------------
__global__ void k_copy(float* dst, const float* src, int n) {
    int i = blockIdx.x * blockDim.x + threadIdx.x;
    if (i < n) dst[i] = src[i];
}
__global__ void k_add(float* dst, const float* a, const float* b, int n) {
    int i = blockIdx.x * blockDim.x + threadIdx.x;
    if (i < n) dst[i] = a[i] + b[i];
}
__global__ void k_posl(float* dst, const float* pos, const float* lev, int n) {
    int i = blockIdx.x * blockDim.x + threadIdx.x;
    if (i >= n) return;
    int d = i % DD;
    int t = (i / DD) % LENT;
    int lvl = (t < 5776) ? 0 : (t < 7220) ? 1 : (t < 7581) ? 2 : 3;
    dst[i] = pos[i] + lev[lvl * DD + d];
}
__global__ void k_addqp(float* dst, const float* out, const float* qe, int n) {
    int i = blockIdx.x * blockDim.x + threadIdx.x;
    if (i >= n) return;
    int d = i % DD;
    int q = (i / DD) % NQQ;
    dst[i] = out[i] + qe[q * (2 * DD) + d];
}
__global__ void k_inittgt(float* dst, const float* qe, int n) {
    int i = blockIdx.x * blockDim.x + threadIdx.x;
    if (i >= n) return;
    int d = i % DD;
    int q = (i / DD) % NQQ;
    dst[i] = qe[q * (2 * DD) + DD + d];
}

// ---------------------------------------------------------------------------
// Softmax over 16 (MSDA attention weights), in place. Thread per row.
// ---------------------------------------------------------------------------
__global__ void k_softmax16(float* p, int rows) {
    int r = blockIdx.x * blockDim.x + threadIdx.x;
    if (r >= rows) return;
    float* x = p + (size_t)r * 16;
    float m = x[0];
    #pragma unroll
    for (int i = 1; i < 16; ++i) m = fmaxf(m, x[i]);
    float e[16], s = 0.0f;
    #pragma unroll
    for (int i = 0; i < 16; ++i) { e[i] = expf(x[i] - m); s += e[i]; }
    float inv = 1.0f / s;
    #pragma unroll
    for (int i = 0; i < 16; ++i) x[i] = e[i] * inv;
}

// ---------------------------------------------------------------------------
// Residual + LayerNorm: y = g*(x+a - mu)/sqrt(var+eps) + b.  Wave per row.
// ---------------------------------------------------------------------------
__global__ void k_ln(const float* __restrict__ x, const float* __restrict__ a,
                     const float* __restrict__ g, const float* __restrict__ be,
                     float* __restrict__ y, int rows)
{
    int row  = blockIdx.x * (blockDim.x >> 5) + (threadIdx.x >> 5);
    int lane = threadIdx.x & 31;
    if (row >= rows) return;
    const float* xr = x + (size_t)row * DD;
    const float* ar = a + (size_t)row * DD;
    float v[8], s = 0.0f;
    #pragma unroll
    for (int i = 0; i < 8; ++i) { v[i] = xr[lane + 32 * i] + ar[lane + 32 * i]; s += v[i]; }
    #pragma unroll
    for (int off = 16; off > 0; off >>= 1) s += __shfl_xor(s, off, 32);
    float mu = s * (1.0f / 256.0f);
    float s2 = 0.0f;
    #pragma unroll
    for (int i = 0; i < 8; ++i) { float d = v[i] - mu; s2 += d * d; }
    #pragma unroll
    for (int off = 16; off > 0; off >>= 1) s2 += __shfl_xor(s2, off, 32);
    float rinv = rsqrtf(s2 * (1.0f / 256.0f) + 1e-5f);
    float* yr = y + (size_t)row * DD;
    #pragma unroll
    for (int i = 0; i < 8; ++i) {
        int d = lane + 32 * i;
        yr[d] = g[d] * (v[i] - mu) * rinv + be[d];
    }
}

// ---------------------------------------------------------------------------
// Encoder reference points (x fastest, (i+0.5)/H grid)
// ---------------------------------------------------------------------------
__global__ void k_refenc(float* ref) {
    int q = blockIdx.x * blockDim.x + threadIdx.x;
    if (q >= LENT) return;
    const int st[4] = {0, 5776, 7220, 7581};
    const int Hs[4] = {76, 38, 19, 10};
    const int Ws[4] = {76, 38, 19, 10};
    int lvl = (q < 5776) ? 0 : (q < 7220) ? 1 : (q < 7581) ? 2 : 3;
    int i = q - st[lvl];
    int Wl = Ws[lvl], Hl = Hs[lvl];
    int yy = i / Wl, xx = i % Wl;
    ref[q * 2 + 0] = (xx + 0.5f) / (float)Wl;
    ref[q * 2 + 1] = (yy + 0.5f) / (float)Hl;
}

// Decoder reference: sigmoid(qp @ ref_w + ref_b).  Thread per (q, c).
__global__ void k_decref(const float* qe, const float* rw_, const float* rb_, float* ref) {
    int i = blockIdx.x * blockDim.x + threadIdx.x;
    if (i >= NQQ * 2) return;
    int c = i & 1, q = i >> 1;
    float s = rb_[c];
    for (int k = 0; k < DD; ++k) s += qe[q * (2 * DD) + k] * rw_[k * 2 + c];
    ref[i] = 1.0f / (1.0f + expf(-s));
}

// ---------------------------------------------------------------------------
// Multi-scale deformable attention sampling.  One wave per (b, q, h),
// lane = channel (HD = 32 = wave32). Wave-uniform ref/off/attn addresses.
// ---------------------------------------------------------------------------
__global__ void k_msda(const float* __restrict__ value, int lenv,
                       const float* __restrict__ off, const float* __restrict__ attn,
                       const float* __restrict__ ref, float* __restrict__ out, int Lq)
{
    int wid  = blockIdx.x * (blockDim.x >> 5) + (threadIdx.x >> 5);
    int lane = threadIdx.x & 31;
    int total = NB * Lq * NHH;
    if (wid >= total) return;
    int h = wid % NHH;
    int q = (wid / NHH) % Lq;
    int b = wid / (NHH * Lq);

    const int st[4] = {0, 5776, 7220, 7581};
    const int Hs[4] = {76, 38, 19, 10};
    const int Ws[4] = {76, 38, 19, 10};

    float rx = ref[q * 2 + 0];
    float ry = ref[q * 2 + 1];
    const float* op = off  + ((size_t)(b * Lq + q) * NHH + h) * 32;   // NL*NP*2
    const float* ap = attn + ((size_t)(b * Lq + q) * NHH + h) * 16;   // NL*NP

    float acc = 0.0f;
    #pragma unroll
    for (int lvl = 0; lvl < 4; ++lvl) {
        int Hl = Hs[lvl], Wl = Ws[lvl];
        const float* vb = value + ((size_t)b * lenv + st[lvl]) * DD + h * HDD + lane;
        #pragma unroll
        for (int p = 0; p < 4; ++p) {
            float ox = op[(lvl * 4 + p) * 2 + 0];
            float oy = op[(lvl * 4 + p) * 2 + 1];
            float aw = ap[lvl * 4 + p];
            float x = rx * (float)Wl + ox - 0.5f;
            float y = ry * (float)Hl + oy - 0.5f;
            float x0 = floorf(x), y0 = floorf(y);
            float s = 0.0f;
            #pragma unroll
            for (int dy = 0; dy < 2; ++dy)
            #pragma unroll
            for (int dx = 0; dx < 2; ++dx) {
                float xi = x0 + (float)dx, yi = y0 + (float)dy;
                float wgt = (1.0f - fabsf(x - xi)) * (1.0f - fabsf(y - yi));
                bool valid = (xi >= 0.0f) && (xi <= (float)(Wl - 1)) &&
                             (yi >= 0.0f) && (yi <= (float)(Hl - 1));
                float xc = fminf(fmaxf(xi, 0.0f), (float)(Wl - 1));
                float yc = fminf(fmaxf(yi, 0.0f), (float)(Hl - 1));
                int idx = (int)(yc * (float)Wl + xc);
                float sv = vb[(size_t)idx * DD];
                s += valid ? (sv * wgt) : 0.0f;
            }
            acc += aw * s;
        }
    }
    out[((size_t)(b * Lq + q) * NHH + h) * HDD + lane] = acc;
}

// ---------------------------------------------------------------------------
// Decoder MHA (tiny: 2 x 8 x 300 x 300): scores / softmax / AV
// qk buffer: [B, NQ, 512]  (cols 0..255 = q heads, 256..511 = k heads)
// ---------------------------------------------------------------------------
__global__ void k_scores(const float* __restrict__ qk, float* __restrict__ sc) {
    int i = blockIdx.x * blockDim.x + threadIdx.x;
    int total = NB * NHH * NQQ * NQQ;
    if (i >= total) return;
    int k  = i % NQQ;
    int qy = (i / NQQ) % NQQ;
    int h  = (i / (NQQ * NQQ)) % NHH;
    int b  = i / (NQQ * NQQ * NHH);
    const float* qp = qk + (size_t)(b * NQQ + qy) * 512 + h * HDD;
    const float* kp = qk + (size_t)(b * NQQ + k) * 512 + 256 + h * HDD;
    float s = 0.0f;
    #pragma unroll
    for (int d = 0; d < HDD; ++d) s += qp[d] * kp[d];
    sc[i] = s * 0.17677669529663687f;   // 1/sqrt(32)
}

__global__ void k_softmax_rows(float* p, int rows, int len) {
    int row  = blockIdx.x * (blockDim.x >> 5) + (threadIdx.x >> 5);
    int lane = threadIdx.x & 31;
    if (row >= rows) return;
    float* x = p + (size_t)row * len;
    float m = -3.0e38f;
    for (int k = lane; k < len; k += 32) m = fmaxf(m, x[k]);
    #pragma unroll
    for (int off = 16; off > 0; off >>= 1) m = fmaxf(m, __shfl_xor(m, off, 32));
    float s = 0.0f;
    for (int k = lane; k < len; k += 32) { float e = expf(x[k] - m); x[k] = e; s += e; }
    #pragma unroll
    for (int off = 16; off > 0; off >>= 1) s += __shfl_xor(s, off, 32);
    float inv = 1.0f / s;
    for (int k = lane; k < len; k += 32) x[k] *= inv;
}

__global__ void k_av(const float* __restrict__ sc, const float* __restrict__ v,
                     float* __restrict__ o) {
    int i = blockIdx.x * blockDim.x + threadIdx.x;
    int total = NB * NQQ * DD;
    if (i >= total) return;
    int d = i % HDD;
    int h = (i / HDD) % NHH;
    int q = (i / DD) % NQQ;
    int b = i / (DD * NQQ);
    const float* ar = sc + ((size_t)(b * NHH + h) * NQQ + q) * NQQ;
    const float* vp = v + (size_t)(b * NQQ) * DD + h * HDD + d;
    float s = 0.0f;
    for (int k = 0; k < NQQ; ++k) s += ar[k] * vp[(size_t)k * DD];
    o[i] = s;
}

// ---------------------------------------------------------------------------
// Host orchestration
// ---------------------------------------------------------------------------
static inline int cdiv(int a, int b) { return (a + b - 1) / b; }

static void gemm(const float* A, int lda, const float* W, int ldw, const float* bias,
                 float* C, int ldc, int M, int N, int K, int relu, hipStream_t s) {
    dim3 g((unsigned)(N / 64), (unsigned)cdiv(M, 128));
    gemm_bf16<<<g, 256, 0, s>>>(A, lda, W, ldw, bias, C, ldc, M, N, K, relu);
}

extern "C" void kernel_launch(void* const* d_in, const int* in_sizes, int n_in,
                              void* d_out, int out_size, void* d_ws, size_t ws_size,
                              hipStream_t stream)
{
    (void)in_sizes; (void)n_in; (void)out_size; (void)ws_size;

    const float* src        = (const float*)d_in[0];
    const float* pos        = (const float*)d_in[1];
    const float* qe         = (const float*)d_in[2];
    const float* lev        = (const float*)d_in[3];
    const float* ref_w      = (const float*)d_in[4];
    const float* ref_b      = (const float*)d_in[5];
    const float* enc_off_w  = (const float*)d_in[6];
    const float* enc_off_b  = (const float*)d_in[7];
    const float* enc_aw_w   = (const float*)d_in[8];
    const float* enc_aw_b   = (const float*)d_in[9];
    const float* enc_val_w  = (const float*)d_in[10];
    const float* enc_val_b  = (const float*)d_in[11];
    const float* enc_out_w  = (const float*)d_in[12];
    const float* enc_out_b  = (const float*)d_in[13];
    const float* enc_ln1    = (const float*)d_in[14];
    const float* enc_l1_w   = (const float*)d_in[15];
    const float* enc_l1_b   = (const float*)d_in[16];
    const float* enc_l2_w   = (const float*)d_in[17];
    const float* enc_l2_b   = (const float*)d_in[18];
    const float* enc_ln2    = (const float*)d_in[19];
    const float* dec_sa_w   = (const float*)d_in[20];
    const float* dec_sa_b   = (const float*)d_in[21];
    const float* dec_sao_w  = (const float*)d_in[22];
    const float* dec_sao_b  = (const float*)d_in[23];
    const float* dec_ln2    = (const float*)d_in[24];
    const float* dec_off_w  = (const float*)d_in[25];
    const float* dec_off_b  = (const float*)d_in[26];
    const float* dec_aw_w   = (const float*)d_in[27];
    const float* dec_aw_b   = (const float*)d_in[28];
    const float* dec_val_w  = (const float*)d_in[29];
    const float* dec_val_b  = (const float*)d_in[30];
    const float* dec_out_w  = (const float*)d_in[31];
    const float* dec_out_b  = (const float*)d_in[32];
    const float* dec_ln1    = (const float*)d_in[33];
    const float* dec_l1_w   = (const float*)d_in[34];
    const float* dec_l1_b   = (const float*)d_in[35];
    const float* dec_l2_w   = (const float*)d_in[36];
    const float* dec_l2_b   = (const float*)d_in[37];
    const float* dec_ln3    = (const float*)d_in[38];

    const int    NT  = NB * LENT;            // 15362 encoder tokens
    const size_t ND  = (size_t)NT * DD;      // 3,932,672 floats
    const int    NTd = NB * NQQ;             // 600 decoder tokens
    const int    nd  = NTd * DD;             // 153,600

    float* w = (float*)d_ws;
    size_t o = 0;
    auto alloc = [&](size_t n) {             // 256B-aligned sub-buffers
        n = (n + 63) & ~(size_t)63;
        float* p = w + o; o += n; return p;
    };

    float* mem    = alloc(ND);
    float* posl   = alloc(ND);
    float* qbuf   = alloc(ND);
    float* val    = alloc(ND);
    float* offb   = alloc(ND);               // OFF = 256
    float* attnb  = alloc((size_t)NT * 128); // AW = 128
    float* msdab  = alloc(ND);
    float* ffb    = alloc((size_t)NT * DFFF);
    float* refenc = alloc((size_t)LENT * 2);
    float* decref = alloc((size_t)NQQ * 2);
    float* dout   = alloc((size_t)nd);
    float* ddq    = alloc((size_t)nd);
    float* dqk    = alloc((size_t)NTd * 512);
    float* dv     = alloc((size_t)nd);
    float* dsa    = alloc((size_t)nd);
    float* doffb  = alloc((size_t)nd);
    float* dattnb = alloc((size_t)NTd * 128);
    float* dmsda  = alloc((size_t)nd);
    float* dffb   = alloc((size_t)NTd * DFFF);
    float* dsc    = alloc((size_t)NB * NHH * NQQ * NQQ);

    const int nE = (int)ND;
    k_copy  <<<cdiv(nE, 256), 256, 0, stream>>>(mem, src, nE);
    k_posl  <<<cdiv(nE, 256), 256, 0, stream>>>(posl, pos, lev, nE);
    k_refenc<<<cdiv(LENT, 256), 256, 0, stream>>>(refenc);

    // ------------------------------ Encoder ------------------------------
    for (int i = 0; i < 6; ++i) {
        const float* vw  = enc_val_w + (size_t)i * DD * DD;
        const float* vb  = enc_val_b + (size_t)i * DD;
        const float* ow  = enc_off_w + (size_t)i * DD * 256;
        const float* ob  = enc_off_b + (size_t)i * 256;
        const float* aww = enc_aw_w  + (size_t)i * DD * 128;
        const float* awb = enc_aw_b  + (size_t)i * 128;
        const float* pw  = enc_out_w + (size_t)i * DD * DD;
        const float* pb  = enc_out_b + (size_t)i * DD;

        k_add<<<cdiv(nE, 256), 256, 0, stream>>>(qbuf, mem, posl, nE);   // q = mem + pos
        gemm(mem,  DD, vw,  DD,  vb,  val,   DD,  NT, DD,  DD, 0, stream);
        gemm(qbuf, DD, ow,  256, ob,  offb,  256, NT, 256, DD, 0, stream);
        gemm(qbuf, DD, aww, 128, awb, attnb, 128, NT, 128, DD, 0, stream);
        k_softmax16<<<cdiv(NT * NHH, 256), 256, 0, stream>>>(attnb, NT * NHH);
        {
            int waves = NB * LENT * NHH;
            k_msda<<<cdiv(waves, 8), 256, 0, stream>>>(val, LENT, offb, attnb,
                                                       refenc, msdab, LENT);
        }
        gemm(msdab, DD, pw, DD, pb, qbuf, DD, NT, DD, DD, 0, stream);
        k_ln<<<cdiv(NT, 8), 256, 0, stream>>>(mem, qbuf,
            enc_ln1 + (size_t)i * 512, enc_ln1 + (size_t)i * 512 + 256, mem, NT);

        gemm(mem, DD, enc_l1_w + (size_t)i * DD * DFFF, DFFF,
             enc_l1_b + (size_t)i * DFFF, ffb, DFFF, NT, DFFF, DD, 1, stream);
        gemm(ffb, DFFF, enc_l2_w + (size_t)i * DFFF * DD, DD,
             enc_l2_b + (size_t)i * DD, qbuf, DD, NT, DD, DFFF, 0, stream);
        k_ln<<<cdiv(NT, 8), 256, 0, stream>>>(mem, qbuf,
            enc_ln2 + (size_t)i * 512, enc_ln2 + (size_t)i * 512 + 256, mem, NT);
    }

    // ------------------------------ Decoder ------------------------------
    k_decref <<<cdiv(NQQ * 2, 256), 256, 0, stream>>>(qe, ref_w, ref_b, decref);
    k_inittgt<<<cdiv(nd, 256), 256, 0, stream>>>(dout, qe, nd);

    for (int i = 0; i < 6; ++i) {
        const float* saw  = dec_sa_w  + (size_t)i * DD * (3 * DD);
        const float* sab  = dec_sa_b  + (size_t)i * (3 * DD);
        const float* saow = dec_sao_w + (size_t)i * DD * DD;
        const float* saob = dec_sao_b + (size_t)i * DD;

        // ---- self attention ----
        k_addqp<<<cdiv(nd, 256), 256, 0, stream>>>(ddq, dout, qe, nd);   // q = out + qp
        gemm(ddq,  DD, saw,       3 * DD, sab,       dqk, 512, NTd, 512, DD, 0, stream);
        gemm(dout, DD, saw + 512, 3 * DD, sab + 512, dv,  DD,  NTd, DD,  DD, 0, stream);
        k_scores<<<cdiv(NB * NHH * NQQ * NQQ, 256), 256, 0, stream>>>(dqk, dsc);
        k_softmax_rows<<<cdiv(NB * NHH * NQQ, 8), 256, 0, stream>>>(dsc, NB * NHH * NQQ, NQQ);
        k_av<<<cdiv(nd, 256), 256, 0, stream>>>(dsc, dv, dsa);
        gemm(dsa, DD, saow, DD, saob, ddq, DD, NTd, DD, DD, 0, stream);
        k_ln<<<cdiv(NTd, 8), 256, 0, stream>>>(dout, ddq,
            dec_ln2 + (size_t)i * 512, dec_ln2 + (size_t)i * 512 + 256, dout, NTd);

        // ---- cross (deformable) attention ----
        k_addqp<<<cdiv(nd, 256), 256, 0, stream>>>(ddq, dout, qe, nd);
        gemm(mem, DD, dec_val_w + (size_t)i * DD * DD, DD,
             dec_val_b + (size_t)i * DD, val, DD, NT, DD, DD, 0, stream);
        gemm(ddq, DD, dec_off_w + (size_t)i * DD * 256, 256,
             dec_off_b + (size_t)i * 256, doffb, 256, NTd, 256, DD, 0, stream);
        gemm(ddq, DD, dec_aw_w + (size_t)i * DD * 128, 128,
             dec_aw_b + (size_t)i * 128, dattnb, 128, NTd, 128, DD, 0, stream);
        k_softmax16<<<cdiv(NTd * NHH, 256), 256, 0, stream>>>(dattnb, NTd * NHH);
        k_msda<<<cdiv(NB * NQQ * NHH, 8), 256, 0, stream>>>(val, LENT, doffb, dattnb,
                                                            decref, dmsda, NQQ);
        gemm(dmsda, DD, dec_out_w + (size_t)i * DD * DD, DD,
             dec_out_b + (size_t)i * DD, ddq, DD, NTd, DD, DD, 0, stream);
        k_ln<<<cdiv(NTd, 8), 256, 0, stream>>>(dout, ddq,
            dec_ln1 + (size_t)i * 512, dec_ln1 + (size_t)i * 512 + 256, dout, NTd);

        // ---- FFN ----
        gemm(dout, DD, dec_l1_w + (size_t)i * DD * DFFF, DFFF,
             dec_l1_b + (size_t)i * DFFF, dffb, DFFF, NTd, DFFF, DD, 1, stream);
        gemm(dffb, DFFF, dec_l2_w + (size_t)i * DFFF * DD, DD,
             dec_l2_b + (size_t)i * DD, ddq, DD, NTd, DD, DFFF, 0, stream);
        float* dst = (i == 5) ? (float*)d_out : dout;
        k_ln<<<cdiv(NTd, 8), 256, 0, stream>>>(dout, ddq,
            dec_ln3 + (size_t)i * 512, dec_ln3 + (size_t)i * 512 + 256, dst, NTd);
    }
}
